// KANCNN_69630009803261
// MI455X (gfx1250) — matrix operand order repaired and checked by reading, not compiled, not run on hardware
//
#include <hip/hip_runtime.h>
#include <hip/hip_bf16.h>
#include <math.h>

// ---------------------------------------------------------------------------
// KAN-CNN forward for MI455X (gfx1250, wave32, WMMA).
// conv2 + fc1: fused f16 WMMA GEMMs (f32 accum), double-buffered LDS with
// async global->LDS staging overlapped with WMMA (partial s_wait_asynccnt),
// and a scheduling fence so all ds_load_b128 fragment loads issue before the
// WMMA burst (partial s_wait_dscnt -> LDS latency hidden under the XDL pipe).
// conv1 is VALU (K=60, too skinny for WMMA). BN affine fused into ReLU+pool.
// Workspace requirement: ~215 MB.
// ---------------------------------------------------------------------------

typedef __attribute__((ext_vector_type(16))) _Float16 v16h;
typedef __attribute__((ext_vector_type(8)))  float    v8f;

#define DEV __device__ __forceinline__

DEV float silu_f(float v) { return v / (1.0f + __expf(-v)); }

// Async global->LDS copy, 16 bytes per lane. GVS mode:
//   LDS[lds_off] = MEM[saddr_base + vgpr_off]
DEV void async_g2l_b128(unsigned lds_off, const void* base, unsigned goff) {
    asm volatile("global_load_async_to_lds_b128 %0, %1, %2"
                 :: "v"(lds_off), "v"(goff), "s"(base)
                 : "memory");
}
template <int N> DEV void wait_async() {
    asm volatile("s_wait_asynccnt %0" :: "i"(N) : "memory");
}
// Scheduling fence: LDS loads may not sink below this point.
DEV void sched_fence() { asm volatile("" ::: "memory"); }

// Cubic B-spline basis (grid_size=1, order=3): uniform grid -7..7 step 2.
DEV void bspline4(float x, float* o) {
    float b0[7];
#pragma unroll
    for (int i = 0; i < 7; i++) {
        float g0 = -7.0f + 2.0f * i;
        b0[i] = (x >= g0 && x < g0 + 2.0f) ? 1.0f : 0.0f;
    }
    float b1[6];
#pragma unroll
    for (int i = 0; i < 6; i++) {
        float gi = -7.0f + 2.0f * i;
        b1[i] = (x - gi) * 0.5f * b0[i] + ((gi + 4.0f) - x) * 0.5f * b0[i + 1];
    }
    float b2[5];
#pragma unroll
    for (int i = 0; i < 5; i++) {
        float gi = -7.0f + 2.0f * i;
        b2[i] = (x - gi) * 0.25f * b1[i] + ((gi + 6.0f) - x) * 0.25f * b1[i + 1];
    }
#pragma unroll
    for (int i = 0; i < 4; i++) {
        float gi = -7.0f + 2.0f * i;
        o[i] = (x - gi) * (1.0f / 6.0f) * b2[i] + ((gi + 8.0f) - x) * (1.0f / 6.0f) * b2[i + 1];
    }
}

// ---------------- init / weight packing ----------------

__global__ void k_zero(float* stats) {
    int t = blockIdx.x * 256 + threadIdx.x;
    if (t < 1024) stats[t] = 0.0f;
}

// N-major packed weights (rows 16B-aligned for async b128 staging):
//   wk2n  [64][640]  conv2 combined (k<128 base/silu, k>=128 spline)
//   wfc1n [64][3136] fc1 weight (same layout as fc1_w, cast to f16)
__global__ void k_pack(const float* __restrict__ bw2, const float* __restrict__ sw2,
                       const float* __restrict__ fc1w,
                       _Float16* __restrict__ wk2n, _Float16* __restrict__ wfc1n) {
    int t = blockIdx.x * 256 + threadIdx.x;
    if (t < 640 * 64) {
        int n = t / 640, k = t % 640;
        float v;
        if (k < 128) v = bw2[n * 128 + k];
        else         v = sw2[n * 512 + (k - 128)];   // (n*128+i)*4+j == n*512+(k-128)
        wk2n[t] = (_Float16)v;
    }
    int t2 = t - 640 * 64;
    if (t2 >= 0 && t2 < 3136 * 64) {
        wfc1n[t2] = (_Float16)fc1w[t2];
    }
}

// ---------------- conv1 (VALU, K=60) + BN1 stats ----------------

__global__ void __launch_bounds__(256)
k_conv1(const float* __restrict__ x, const float* __restrict__ bw1,
        const float* __restrict__ sw1, float* __restrict__ out,
        float* __restrict__ gsum, float* __restrict__ gsq) {
    __shared__ float ssum[32], ssq[32];
    int t = threadIdx.x;
    if (t < 32) { ssum[t] = 0.0f; ssq[t] = 0.0f; }
    __syncthreads();
    int idx = blockIdx.x * 256 + t;               // exactly 1024*31*31 threads
    int b = idx / 961, r = idx % 961, y = r / 31, xx = r % 31;
    float sb[12], bs[48];
#pragma unroll
    for (int c = 0; c < 3; c++)
#pragma unroll
        for (int dy = 0; dy < 2; dy++)
#pragma unroll
            for (int dx = 0; dx < 2; dx++) {
                float vv = x[((b * 3 + c) * 32 + y + dy) * 32 + xx + dx];
                int i = c * 4 + dy * 2 + dx;
                sb[i] = silu_f(vv);
                bspline4(vv, &bs[i * 4]);
            }
    int lane = t & 31;
    for (int o = 0; o < 32; o++) {
        float acc = 0.0f;
#pragma unroll
        for (int i = 0; i < 12; i++) acc += sb[i] * bw1[o * 12 + i];   // uniform -> s_load
#pragma unroll
        for (int i = 0; i < 12; i++)
#pragma unroll
            for (int j = 0; j < 4; j++) acc += bs[i * 4 + j] * sw1[(o * 12 + i) * 4 + j];
        out[((b * 32 + o) * 31 + y) * 31 + xx] = acc;
        float s = acc, q = acc * acc;
        for (int off = 16; off > 0; off >>= 1) {                       // wave32 reduce
            s += __shfl_down(s, off);
            q += __shfl_down(q, off);
        }
        if (lane == 0) { atomicAdd(&ssum[o], s); atomicAdd(&ssq[o], q); }
    }
    __syncthreads();
    if (t < 32) { atomicAdd(&gsum[t], ssum[t]); atomicAdd(&gsq[t], ssq[t]); }
}

// ---------------- BN finalize (sum/sq -> scale/shift) ----------------

__global__ void k_bnfin(const float* __restrict__ sum, const float* __restrict__ sq,
                        const float* __restrict__ g, const float* __restrict__ bta,
                        float* __restrict__ scale, float* __restrict__ shift,
                        int C, float invN) {
    int c = threadIdx.x;
    if (c < C) {
        float m = sum[c] * invN;
        float var = sq[c] * invN - m * m;
        float s = g[c] * rsqrtf(var + 1e-5f);
        scale[c] = s;
        shift[c] = bta[c] - m * s;
    }
}

// ---------------- fused BN1-affine + ReLU + maxpool2 ----------------

__global__ void __launch_bounds__(256)
k_pool1(const float* __restrict__ cin, const float* __restrict__ sc,
        const float* __restrict__ sh, float* __restrict__ pout) {
    int idx = blockIdx.x * 256 + threadIdx.x;     // exactly 1024*32*15*15
    int b = idx / (32 * 225), r = idx % (32 * 225);
    int c = r / 225, rr = r % 225, y = rr / 15, xx = rr % 15;
    float s = sc[c], h = sh[c];
    const float* base = cin + ((b * 32 + c) * 31 + 2 * y) * 31 + 2 * xx;
    float mval = 0.0f;                            // relu => all candidates >= 0
#pragma unroll
    for (int dy = 0; dy < 2; dy++)
#pragma unroll
        for (int dx = 0; dx < 2; dx++)
            mval = fmaxf(mval, base[dy * 31 + dx] * s + h);
    pout[((b * 32 + c) * 15 + y) * 15 + xx] = mval;
}

// ---------------- shared WMMA tile pass ----------------
// As: [128][72] f16 row-major (M x Kchunk).  Bst: [64][72] f16 TRANSPOSED
// (N x Kchunk). ALL fragment ds_load_b128s issue before the WMMA burst
// (sched_fence), loads ordered in consumption order -> partial dscnt waits.

DEV void wmma_chunk(const _Float16 (*As)[72], const _Float16 (*Bst)[72],
                    int w, int half, int l16, v8f acc[4]) {
    v16h a[2], bf[2][4];
#pragma unroll
    for (int ks = 0; ks < 2; ks++) {
        const _Float16* ap = &As[(w << 4) + l16][ks * 32 + half * 8];
#pragma unroll
        for (int j = 0; j < 8; j++) {             // A frag: 2x16B contiguous
            a[ks][j]     = ap[j];
            a[ks][8 + j] = ap[16 + j];
        }
    }
#pragma unroll
    for (int ks = 0; ks < 2; ks++)
#pragma unroll
        for (int nt = 0; nt < 4; nt++) {          // loads in exact use order
            const _Float16* bp = &Bst[nt * 16 + l16][ks * 32 + half * 8];
#pragma unroll
            for (int j = 0; j < 8; j++) { bf[ks][nt][j] = bp[j]; bf[ks][nt][8 + j] = bp[16 + j]; }  // 2x16B contiguous
        }
    sched_fence();                                // keep loads above the MMA burst
#pragma unroll
    for (int ks = 0; ks < 2; ks++)
#pragma unroll
        for (int nt = 0; nt < 4; nt++)
            acc[nt] = __builtin_amdgcn_wmma_f32_16x16x32_f16(
                false, a[ks], false, bf[ks][nt], (short)0, acc[nt], false, false);
}

// dynamic LDS layout (double-buffered):
//   As0 @0, As1 @18432, Bst0 @36864, Bst1 @46080, stats @55296  (55808 B)
#define SM_AS(buf)  ((unsigned)(buf) * 18432u)
#define SM_BS(buf)  (36864u + (unsigned)(buf) * 9216u)
#define SM_ST_OFF   55296u
#define SM_BYTES    55808u

// ---------------- conv2: fused feature-gen + WMMA GEMM + BN2 stats -----------
// M=200704 patches (128/block), N=64, K=640 (10 chunks of 64 = 2 wmma k-steps).

DEV void conv2_issue_b(const _Float16* wk2n, int kc, unsigned bsoff, int t) {
#pragma unroll
    for (int it = 0; it < 2; it++) {              // 64 rows x 8 segs = 512
        int seg = t + it * 256;
        int n = seg >> 3, kseg = seg & 7;
        async_g2l_b128(bsoff + (unsigned)(n * 144 + kseg * 16), wk2n,
                       (unsigned)((n * 640 + kc * 64 + kseg * 8) * 2));
    }
}

DEV void conv2_gen_a(const float* __restrict__ p1, int kc, int m0,
                     _Float16 (*As)[72], int t) {
    if (kc < 2) {                                  // base branch: silu features
        for (int e = t; e < 8192; e += 256) {
            int m = e >> 6, kl = e & 63, k = kc * 64 + kl;
            int mg = m0 + m, b = mg / 196, rr = mg % 196, y = rr / 14, xx = rr % 14;
            int c = k >> 2, dy = (k >> 1) & 1, dx = k & 1;
            float vv = p1[((b * 32 + c) * 15 + y + dy) * 15 + xx + dx];
            As[m][kl] = (_Float16)silu_f(vv);
        }
    } else {                                       // spline branch: 4 bases/input
        for (int e = t; e < 2048; e += 256) {
            int m = e >> 4, il = e & 15;
            int i = (kc - 2) * 16 + il;
            int mg = m0 + m, b = mg / 196, rr = mg % 196, y = rr / 14, xx = rr % 14;
            int c = i >> 2, dy = (i >> 1) & 1, dx = i & 1;
            float vv = p1[((b * 32 + c) * 15 + y + dy) * 15 + xx + dx];
            float bsv[4];
            bspline4(vv, bsv);
#pragma unroll
            for (int j = 0; j < 4; j++) As[m][il * 4 + j] = (_Float16)bsv[j];
        }
    }
}

__global__ void __launch_bounds__(256)
k_conv2(const float* __restrict__ p1, const _Float16* __restrict__ wk2n,
        float* __restrict__ out, float* __restrict__ gsum, float* __restrict__ gsq) {
    extern __shared__ __align__(16) char smem[];
    float* ssum = (float*)(smem + SM_ST_OFF);
    float* ssq  = ssum + 64;
    int t = threadIdx.x;
    if (t < 64) { ssum[t] = 0.0f; ssq[t] = 0.0f; }
    int m0 = blockIdx.x * 128;
    int w = t >> 5, lane = t & 31, half = lane >> 4, l16 = lane & 15;
    v8f acc[4] = {};
    // prologue: stage chunk 0 into buffer 0
    conv2_issue_b(wk2n, 0, SM_BS(0), t);
    conv2_gen_a(p1, 0, m0, (_Float16(*)[72])(smem + SM_AS(0)), t);
    for (int kc = 0; kc < 10; kc++) {
        int cur = kc & 1, nxt = cur ^ 1;
        __syncthreads();                           // readers of buf[nxt] done
        if (kc + 1 < 10) {
            conv2_issue_b(wk2n, kc + 1, SM_BS(nxt), t);
            conv2_gen_a(p1, kc + 1, m0, (_Float16(*)[72])(smem + SM_AS(nxt)), t);
            wait_async<2>();                       // chunk kc resident; kc+1 in flight
        } else {
            wait_async<0>();
        }
        __syncthreads();                           // publish buf[cur] to all waves
        wmma_chunk((const _Float16(*)[72])(smem + SM_AS(cur)),
                   (const _Float16(*)[72])(smem + SM_BS(cur)), w, half, l16, acc);
    }
    // epilogue: scatter to [B,64,14,14] + BN2 stats
#pragma unroll
    for (int nt = 0; nt < 4; nt++) {
        int n = nt * 16 + l16;
        float s = 0.0f, q = 0.0f;
#pragma unroll
        for (int j = 0; j < 8; j++) {
            int ml = (w << 4) + half * 8 + j;      // C/D layout: vgpr j -> M=j(+8)
            int mg = m0 + ml, b = mg / 196, rr = mg % 196, y = rr / 14, xx = rr % 14;
            float vv = acc[nt][j];
            out[((b * 64 + n) * 14 + y) * 14 + xx] = vv;
            s += vv; q += vv * vv;
        }
        atomicAdd(&ssum[n], s);
        atomicAdd(&ssq[n], q);
    }
    __syncthreads();
    if (t < 64) { atomicAdd(&gsum[t], ssum[t]); atomicAdd(&gsq[t], ssq[t]); }
}

// ------- fused BN2-affine + ReLU + maxpool2 + flatten -> f16 features -------

__global__ void __launch_bounds__(256)
k_pool2(const float* __restrict__ cin, const float* __restrict__ sc,
        const float* __restrict__ sh, _Float16* __restrict__ feat) {
    int idx = blockIdx.x * 256 + threadIdx.x;     // exactly 1024*64*49
    int b = idx / (64 * 49), r = idx % (64 * 49);
    int c = r / 49, rr = r % 49, y = rr / 7, xx = rr % 7;
    float s = sc[c], h = sh[c];
    const float* base = cin + ((b * 64 + c) * 14 + 2 * y) * 14 + 2 * xx;
    float mval = 0.0f;
#pragma unroll
    for (int dy = 0; dy < 2; dy++)
#pragma unroll
        for (int dx = 0; dx < 2; dx++)
            mval = fmaxf(mval, base[dy * 14 + dx] * s + h);
    feat[b * 3136 + c * 49 + y * 7 + xx] = (_Float16)mval;
}

// ---------------- fc1: WMMA GEMM [1024x3136]x[3136x64] + bias + BN3 stats ----
// A and B tiles async-staged, double-buffered, overlapped with WMMA.

DEV void fc1_issue(const _Float16* feat, const _Float16* wfc1n, int kc, int m0,
                   unsigned asoff, unsigned bsoff, int t) {
#pragma unroll
    for (int it = 0; it < 4; it++) {               // A: 128 rows x 8 segs = 1024
        int seg = t + it * 256;
        int m = seg >> 3, kseg = seg & 7;
        async_g2l_b128(asoff + (unsigned)(m * 144 + kseg * 16), feat,
                       (unsigned)(((m0 + m) * 3136 + kc * 64 + kseg * 8) * 2));
    }
#pragma unroll
    for (int it = 0; it < 2; it++) {               // B: 64 rows x 8 segs = 512
        int seg = t + it * 256;
        int n = seg >> 3, kseg = seg & 7;
        async_g2l_b128(bsoff + (unsigned)(n * 144 + kseg * 16), wfc1n,
                       (unsigned)((n * 3136 + kc * 64 + kseg * 8) * 2));
    }
}

__global__ void __launch_bounds__(256)
k_fc1(const _Float16* __restrict__ feat, const _Float16* __restrict__ wfc1n,
      const float* __restrict__ bias, float* __restrict__ out,
      float* __restrict__ gsum, float* __restrict__ gsq) {
    extern __shared__ __align__(16) char smem[];
    float* ssum = (float*)(smem + SM_ST_OFF);
    float* ssq  = ssum + 64;
    int t = threadIdx.x;
    if (t < 64) { ssum[t] = 0.0f; ssq[t] = 0.0f; }
    int m0 = blockIdx.x * 128;
    int w = t >> 5, lane = t & 31, half = lane >> 4, l16 = lane & 15;
    v8f acc[4] = {};
    fc1_issue(feat, wfc1n, 0, m0, SM_AS(0), SM_BS(0), t);
    for (int kc = 0; kc < 49; kc++) {
        int cur = kc & 1, nxt = cur ^ 1;
        __syncthreads();
        if (kc + 1 < 49) {
            fc1_issue(feat, wfc1n, kc + 1, m0, SM_AS(nxt), SM_BS(nxt), t);
            if (kc + 2 < 49)                       // gfx1250 global_prefetch_b8
                __builtin_prefetch(&feat[(size_t)(m0 + (t >> 1)) * 3136 +
                                         (kc + 2) * 64 + ((t & 1) << 5)], 0, 1);
            wait_async<6>();                       // chunk kc resident; kc+1 in flight
        } else {
            wait_async<0>();
        }
        __syncthreads();
        wmma_chunk((const _Float16(*)[72])(smem + SM_AS(cur)),
                   (const _Float16(*)[72])(smem + SM_BS(cur)), w, half, l16, acc);
    }
#pragma unroll
    for (int nt = 0; nt < 4; nt++) {
        int n = nt * 16 + l16;
        float bv = bias[n];
        float s = 0.0f, q = 0.0f;
#pragma unroll
        for (int j = 0; j < 8; j++) {
            int m = m0 + (w << 4) + half * 8 + j;
            float vv = acc[nt][j] + bv;
            out[m * 64 + n] = vv;
            s += vv; q += vv * vv;
        }
        atomicAdd(&ssum[n], s);
        atomicAdd(&ssq[n], q);
    }
    __syncthreads();
    if (t < 64) { atomicAdd(&gsum[t], ssum[t]); atomicAdd(&gsq[t], ssq[t]); }
}

// ---------------- fc2 (tiny, VALU) with fused BN3+ReLU ----------------

__global__ void k_fc2(const float* __restrict__ h, const float* __restrict__ sc,
                      const float* __restrict__ sh, const float* __restrict__ wt,
                      const float* __restrict__ bias, float* __restrict__ out) {
    int idx = blockIdx.x * 256 + threadIdx.x;     // 1024*10 outputs
    if (idx >= 10240) return;
    int b = idx / 10, o = idx % 10;
    float acc = bias[o];
#pragma unroll 8
    for (int c = 0; c < 64; c++) {
        float v = fmaxf(0.0f, h[b * 64 + c] * sc[c] + sh[c]);
        acc += v * wt[o * 64 + c];
    }
    out[idx] = acc;
}

// ---------------------------------------------------------------------------

extern "C" void kernel_launch(void* const* d_in, const int* in_sizes, int n_in,
                              void* d_out, int out_size, void* d_ws, size_t ws_size,
                              hipStream_t stream) {
    const float* x    = (const float*)d_in[0];
    const float* bw1  = (const float*)d_in[1];
    const float* sw1  = (const float*)d_in[2];
    const float* bn1g = (const float*)d_in[3];
    const float* bn1b = (const float*)d_in[4];
    const float* bw2  = (const float*)d_in[5];
    const float* sw2  = (const float*)d_in[6];
    const float* bn2g = (const float*)d_in[7];
    const float* bn2b = (const float*)d_in[8];
    const float* fc1w = (const float*)d_in[9];
    const float* fc1b = (const float*)d_in[10];
    const float* bn3g = (const float*)d_in[11];
    const float* bn3b = (const float*)d_in[12];
    const float* fc2w = (const float*)d_in[13];
    const float* fc2b = (const float*)d_in[14];
    float* out = (float*)d_out;

    // workspace carve-out
    char* ws = (char*)d_ws;
    size_t off = 0;
    auto carve = [&](size_t bytes) -> char* {
        char* p = ws + off;
        off = (off + bytes + 255) & ~(size_t)255;
        return p;
    };
    float*     stats = (float*)carve(1024 * 4);
    _Float16*  wk2n  = (_Float16*)carve((size_t)640 * 64 * 2);
    _Float16*  wfc1n = (_Float16*)carve((size_t)3136 * 64 * 2);
    float*     c1    = (float*)carve((size_t)1024 * 32 * 31 * 31 * 4);
    float*     p1    = (float*)carve((size_t)1024 * 32 * 15 * 15 * 4);
    float*     c2    = (float*)carve((size_t)1024 * 64 * 14 * 14 * 4);
    _Float16*  feat  = (_Float16*)carve((size_t)1024 * 3136 * 2);
    float*     h1    = (float*)carve((size_t)1024 * 64 * 4);

    float *s1s = stats +   0, *s1q = stats +  32, *sc1 = stats +  64, *sh1 = stats +  96;
    float *s2s = stats + 128, *s2q = stats + 192, *sc2 = stats + 256, *sh2 = stats + 320;
    float *s3s = stats + 384, *s3q = stats + 448, *sc3 = stats + 512, *sh3 = stats + 576;

    k_zero<<<4, 256, 0, stream>>>(stats);
    k_pack<<<944, 256, 0, stream>>>(bw2, sw2, fc1w, wk2n, wfc1n);

    k_conv1<<<3844, 256, 0, stream>>>(x, bw1, sw1, c1, s1s, s1q);
    k_bnfin<<<1, 64, 0, stream>>>(s1s, s1q, bn1g, bn1b, sc1, sh1, 32,
                                  1.0f / (1024.0f * 961.0f));
    k_pool1<<<28800, 256, 0, stream>>>(c1, sc1, sh1, p1);

    k_conv2<<<1568, 256, SM_BYTES, stream>>>(p1, wk2n, c2, s2s, s2q);
    k_bnfin<<<1, 64, 0, stream>>>(s2s, s2q, bn2g, bn2b, sc2, sh2, 64,
                                  1.0f / (1024.0f * 196.0f));
    k_pool2<<<12544, 256, 0, stream>>>(c2, sc2, sh2, feat);

    k_fc1<<<8, 256, SM_BYTES, stream>>>(feat, wfc1n, fc1b, h1, s3s, s3q);
    k_bnfin<<<1, 64, 0, stream>>>(s3s, s3q, bn3g, bn3b, sc3, sh3, 64,
                                  1.0f / 1024.0f);
    k_fc2<<<40, 256, 0, stream>>>(h1, sc3, sh3, fc2w, fc2b, out);
}
